// ExpertChoiceFF_41274635715088
// MI455X (gfx1250) — matrix operand order repaired and checked by reading, not compiled, not run on hardware
//
#include <hip/hip_runtime.h>

// ---------------------------------------------------------------------------
// Expert-choice MoE FFN for MI455X (gfx1250, wave32, WMMA).
// B=2, S=2048, D=1024, E=16, topk=256.
// Heavy GEMMs run on v_wmma_f32_16x16x32_bf16 with f32 accumulation.
// Each wave holds a 32x128 accumulator stripe: every B tile register is fed
// into two WMMAs (two M-subtiles), halving weight traffic from L2.
// ---------------------------------------------------------------------------

typedef __attribute__((ext_vector_type(16))) __bf16 v16bf;
typedef __attribute__((ext_vector_type(8)))  __bf16 v8bf;
typedef __attribute__((ext_vector_type(8)))  float  v8f;

constexpr int Bc  = 2;
constexpr int Sc  = 2048;
constexpr int Dc  = 1024;
constexpr int Ec  = 16;
constexpr int TKc = 256;

__device__ __forceinline__ unsigned short f2bf(float f) {
  union { float f; unsigned u; } v; v.f = f;
  unsigned r = v.u + 0x7FFFu + ((v.u >> 16) & 1u);   // round-to-nearest-even
  return (unsigned short)(r >> 16);
}

// ---------------------------------------------------------------------------
__global__ void k_zero(float* __restrict__ p, int n) {
  int i = blockIdx.x * blockDim.x + threadIdx.x;
  if (i < n) p[i] = 0.0f;
}

// logits[b,e,s] = sum_d x[b,s,d] * gate[d,e]   (stored expert-major for topk)
__global__ __launch_bounds__(256) void k_logits(const float* __restrict__ x,
                                                const float* __restrict__ gate,
                                                float* __restrict__ probsT) {
  const int bs   = blockIdx.x;            // b*S + s
  const int e    = threadIdx.x & 15;
  const int part = threadIdx.x >> 4;      // 16 partial-sum groups
  const float* xr = x + (size_t)bs * Dc;
  const int span = Dc / 16;               // 64
  const int d0   = part * span;
  float acc = 0.f;
#pragma unroll 4
  for (int i = 0; i < span; ++i)
    acc += xr[d0 + i] * gate[(d0 + i) * Ec + e];
  __shared__ float red[16][17];
  red[part][e] = acc;
  __syncthreads();
  if (part == 0) {
    float t = 0.f;
#pragma unroll
    for (int i = 0; i < 16; ++i) t += red[i][e];
    const int b = bs >> 11, s = bs & (Sc - 1);
    probsT[((size_t)(b * Ec + e)) * Sc + s] = t;
  }
}

// softmax over the token axis, in place, one block per (b,e)
__global__ __launch_bounds__(256) void k_softmax(float* __restrict__ probsT) {
  float* row = probsT + (size_t)blockIdx.x * Sc;
  const int tid = threadIdx.x;
  __shared__ float red[256];
  float m = -3.4e38f;
  for (int i = tid; i < Sc; i += 256) m = fmaxf(m, row[i]);
  red[tid] = m; __syncthreads();
  for (int o = 128; o > 0; o >>= 1) {
    if (tid < o) red[tid] = fmaxf(red[tid], red[tid + o]);
    __syncthreads();
  }
  m = red[0]; __syncthreads();
  float s = 0.f;
  for (int i = tid; i < Sc; i += 256) s += __expf(row[i] - m);
  red[tid] = s; __syncthreads();
  for (int o = 128; o > 0; o >>= 1) {
    if (tid < o) red[tid] += red[tid + o];
    __syncthreads();
  }
  const float inv = 1.0f / red[0];
  for (int i = tid; i < Sc; i += 256) row[i] = __expf(row[i] - m) * inv;
}

// top-256 of 2048 per (b,e): bitonic sort of packed (value_bits<<32 | ~index)
// keys in LDS; descending order reproduces jax.lax.top_k (lowest index wins ties).
__global__ __launch_bounds__(1024) void k_topk(const float* __restrict__ probsT,
                                               float* __restrict__ gsel,
                                               int* __restrict__ isel) {
  __shared__ unsigned long long keys[Sc];
  const float* row = probsT + (size_t)blockIdx.x * Sc;
  const int tid = threadIdx.x;
  for (int i = tid; i < Sc; i += 1024) {
    union { float f; unsigned u; } v; v.f = row[i];   // softmax output > 0
    keys[i] = ((unsigned long long)v.u << 32) | (unsigned)(0xFFFFFFFFu - (unsigned)i);
  }
  __syncthreads();
  for (int k = 2; k <= Sc; k <<= 1) {
    for (int j = k >> 1; j > 0; j >>= 1) {
      for (int t = tid; t < Sc; t += 1024) {
        const int l = t ^ j;
        if (l > t) {
          const unsigned long long a = keys[t], c = keys[l];
          const bool asc = (t & k) != 0;               // overall descending
          if (asc ? (a > c) : (a < c)) { keys[t] = c; keys[l] = a; }
        }
      }
      __syncthreads();
    }
  }
  if (tid < TKc) {
    const unsigned long long kk = keys[tid];
    union { unsigned u; float f; } v; v.u = (unsigned)(kk >> 32);
    gsel[(size_t)blockIdx.x * TKc + tid] = v.f;
    isel[(size_t)blockIdx.x * TKc + tid] = (int)(0xFFFFFFFFu - (unsigned)(kk & 0xFFFFFFFFu));
  }
}

// WT[e][n][k] = bf16(W[e][k][n]) — transposed so WMMA B-operand reads are
// contiguous 16-element runs per lane.
__global__ __launch_bounds__(256) void k_w_to_bf16T(const float* __restrict__ W,
                                                    unsigned short* __restrict__ WT) {
  const size_t i = (size_t)blockIdx.x * 256 + threadIdx.x;   // over E*D*D
  const int e = (int)(i >> 20);
  const int k = (int)((i >> 10) & 1023);
  const int n = (int)(i & 1023);
  WT[((size_t)e << 20) | ((size_t)n << 10) | (size_t)k] = f2bf(W[i]);
}

// ---------------------------------------------------------------------------
// Fused FFN: one block per (b, e, 32-row stripe of the 256 selected tokens).
// 8 wave32; each wave owns a 32x128 f32 accumulator stripe (2x8 WMMA tiles),
// reusing each B-operand register for two WMMAs.
// ---------------------------------------------------------------------------
__global__ __launch_bounds__(256) void k_moe_ffn(
    const float* __restrict__ x,
    const float* __restrict__ b1,
    const float* __restrict__ b2,
    const unsigned short* __restrict__ W1T,
    const unsigned short* __restrict__ W2T,
    const float* __restrict__ gsel,
    const int* __restrict__ isel,
    float* __restrict__ y) {
  constexpr int MT = 32;                               // rows per block
  __shared__ __align__(32) unsigned short xs[MT][Dc];  // gathered tokens, bf16 (64KB)
  __shared__ __align__(32) unsigned short hs[MT][Dc];  // relu hidden, bf16   (64KB)
  __shared__ int   tok[MT];
  __shared__ float gv[MT];

  const int bi     = blockIdx.x;
  const int stripe = bi & 7;          // 8 stripes of 32 rows
  const int e      = (bi >> 3) & 15;
  const int b      = bi >> 7;
  const int tid    = threadIdx.x;
  const int lane   = tid & 31;
  const int w      = tid >> 5;
  const int m0     = stripe * MT;
  const size_t be  = (size_t)(b * Ec + e);

  if (tid < MT) {
    tok[tid] = isel[be * TKc + m0 + tid];
    gv[tid]  = gsel[be * TKc + m0 + tid];
  }
  __syncthreads();

  { // gather 32 token rows, f32 -> bf16 (coalesced 64B chunks per 16-thread group)
    const int r0 = tid >> 4, c0 = tid & 15;
#pragma unroll
    for (int rr = 0; rr < 2; ++rr) {
      const int r = r0 + rr * 16;
      const float* src = x + ((size_t)b * Sc + (size_t)tok[r]) * Dc;
#pragma unroll 8
      for (int c = c0; c < Dc; c += 16) xs[r][c] = f2bf(src[c]);
    }
  }
  __syncthreads();

  const int n0    = w * 128;       // this wave's N span
  const int col   = lane & 15;     // A-row index for A loads; N-column for B/C/D
  const int khalf = lane >> 4;

  v8f acc[2][8];
#pragma unroll
  for (int s2 = 0; s2 < 2; ++s2)
#pragma unroll
    for (int i = 0; i < 8; ++i)
#pragma unroll
      for (int j = 0; j < 8; ++j) acc[s2][i][j] = 0.f;

  // ---- GEMM1: h = relu(xin @ W1 + b1) --------------------------------------
  for (int kb = 0; kb < Dc; kb += 32) {
    // A 16x32 bf16 layout: lane half 0 -> K{0..15}, half 1 -> K{16..31} (matches B)
    v16bf a[2];
#pragma unroll
    for (int s2 = 0; s2 < 2; ++s2) {
      v8bf lo = *(const v8bf*)&xs[s2 * 16 + col][kb + khalf * 16];
      v8bf hi = *(const v8bf*)&xs[s2 * 16 + col][kb + khalf * 16 + 8];
#pragma unroll
      for (int i = 0; i < 8; ++i) { a[s2][i] = lo[i]; a[s2][8 + i] = hi[i]; }
    }
    // B 32x16: lane holds column N, contiguous K run (half 0: kb..kb+15, half 1: +16)
    const unsigned short* wb = W1T + ((size_t)e << 20) + (size_t)(kb + khalf * 16);
#pragma unroll
    for (int nt = 0; nt < 8; ++nt) {
      const int n = n0 + nt * 16 + col;
      v16bf bm = *(const v16bf*)(wb + (size_t)n * Dc);
      acc[0][nt] = __builtin_amdgcn_wmma_f32_16x16x32_bf16(
          false, a[0], false, bm, (short)0, acc[0][nt], false, false);
      acc[1][nt] = __builtin_amdgcn_wmma_f32_16x16x32_bf16(
          false, a[1], false, bm, (short)0, acc[1][nt], false, false);
    }
  }
  { // epilogue: bias + relu, spill to LDS as bf16; C/D layout M = j + 8*khalf, N = col
    const int mb = 8 * khalf;
#pragma unroll
    for (int s2 = 0; s2 < 2; ++s2)
#pragma unroll
      for (int nt = 0; nt < 8; ++nt) {
        const int n = n0 + nt * 16 + col;
        const float bb = b1[(size_t)e * Dc + n];
#pragma unroll
        for (int j = 0; j < 8; ++j) {
          float v = acc[s2][nt][j] + bb;
          v = v > 0.f ? v : 0.f;
          hs[s2 * 16 + mb + j][n] = f2bf(v);
          acc[s2][nt][j] = 0.f;
        }
      }
  }
  __syncthreads();

  // ---- GEMM2: out = h @ W2 + b2 --------------------------------------------
  for (int kb = 0; kb < Dc; kb += 32) {
    v16bf a[2];
#pragma unroll
    for (int s2 = 0; s2 < 2; ++s2) {
      v8bf lo = *(const v8bf*)&hs[s2 * 16 + col][kb + khalf * 16];
      v8bf hi = *(const v8bf*)&hs[s2 * 16 + col][kb + khalf * 16 + 8];
#pragma unroll
      for (int i = 0; i < 8; ++i) { a[s2][i] = lo[i]; a[s2][8 + i] = hi[i]; }
    }
    const unsigned short* wb = W2T + ((size_t)e << 20) + (size_t)(kb + khalf * 16);
#pragma unroll
    for (int nt = 0; nt < 8; ++nt) {
      const int n = n0 + nt * 16 + col;
      v16bf bm = *(const v16bf*)(wb + (size_t)n * Dc);
      acc[0][nt] = __builtin_amdgcn_wmma_f32_16x16x32_bf16(
          false, a[0], false, bm, (short)0, acc[0][nt], false, false);
      acc[1][nt] = __builtin_amdgcn_wmma_f32_16x16x32_bf16(
          false, a[1], false, bm, (short)0, acc[1][nt], false, false);
    }
  }
  { // epilogue: bias, gate weight, scatter-add into y
    const int mb = 8 * khalf;
#pragma unroll
    for (int s2 = 0; s2 < 2; ++s2)
#pragma unroll
      for (int nt = 0; nt < 8; ++nt) {
        const int n = n0 + nt * 16 + col;
        const float bb = b2[n];
#pragma unroll
        for (int j = 0; j < 8; ++j) {
          const int m = s2 * 16 + mb + j;
          const float v = (acc[s2][nt][j] + bb) * gv[m];
          unsafeAtomicAdd(&y[((size_t)b * Sc + (size_t)tok[m]) * Dc + n], v);
        }
      }
  }
}

// ---------------------------------------------------------------------------
extern "C" void kernel_launch(void* const* d_in, const int* in_sizes, int n_in,
                              void* d_out, int out_size, void* d_ws, size_t ws_size,
                              hipStream_t stream) {
  (void)in_sizes; (void)n_in; (void)out_size; (void)ws_size;
  const float* x    = (const float*)d_in[0];
  const float* gate = (const float*)d_in[1];
  const float* W1   = (const float*)d_in[2];
  const float* b1   = (const float*)d_in[3];
  const float* W2   = (const float*)d_in[4];
  const float* b2   = (const float*)d_in[5];
  float* y = (float*)d_out;

  // workspace layout (~67.4 MB)
  float* probsT = (float*)d_ws;                                  // B*E*S f32
  float* gsel   = probsT + (size_t)Bc * Ec * Sc;                 // B*E*TK f32
  int*   isel   = (int*)(gsel + (size_t)Bc * Ec * TKc);          // B*E*TK i32
  unsigned short* W1T = (unsigned short*)(isel + (size_t)Bc * Ec * TKc); // E*D*D bf16
  unsigned short* W2T = W1T + (size_t)Ec * Dc * Dc;                      // E*D*D bf16

  const int ytot = Bc * Sc * Dc;
  k_zero<<<(ytot + 255) / 256, 256, 0, stream>>>(y, ytot);
  k_logits<<<Bc * Sc, 256, 0, stream>>>(x, gate, probsT);
  k_softmax<<<Bc * Ec, 256, 0, stream>>>(probsT);
  k_topk<<<Bc * Ec, 1024, 0, stream>>>(probsT, gsel, isel);
  const int wtot = Ec * Dc * Dc;
  k_w_to_bf16T<<<wtot / 256, 256, 0, stream>>>(W1, W1T);
  k_w_to_bf16T<<<wtot / 256, 256, 0, stream>>>(W2, W2T);
  k_moe_ffn<<<Bc * Ec * (TKc / 32), 256, 0, stream>>>(x, b1, b2, W1T, W2T, gsel, isel, y);
}
